// DualLSTMModel_45629732552713
// MI455X (gfx1250) — matrix operand order repaired and checked by reading, not compiled, not run on hardware
//
#include <hip/hip_runtime.h>
#include <stddef.h>

// ---------------------------------------------------------------------------
// Problem constants (from reference)
// ---------------------------------------------------------------------------
#define BB   16
#define TT   128        // T_IN == T_OUT
#define EE   512
#define HH   512
#define G4H  2048       // 4*H
#define VV   32000
#define BT   2048       // B*T

typedef __attribute__((ext_vector_type(16))) __bf16 v16bf;
typedef __attribute__((ext_vector_type(8)))  __bf16 v8bfv;
typedef __attribute__((ext_vector_type(8)))  float  v8f;
typedef __attribute__((ext_vector_type(4)))  unsigned v4u;
typedef __attribute__((ext_vector_type(8)))  int    v8i;
typedef __attribute__((ext_vector_type(4)))  int    v4i;

// ---------------------------------------------------------------------------
// WMMA helpers (V_WMMA_F32_16X16X32_BF16 lane layouts as used here; A/B k order is this code's choice)
//  A (16x32 bf16): lane l: row m=l&15; K = k0 + (l>>4)*16 + {0..15} contiguous
//  B (32x16 bf16): lane l: col n=l&15; K = k0 + (l>>4)*16 + {0..15} contiguous
//  C/D (16x16 f32): lane l, vgpr r: M=(l>>4)*8+r, N=l&15
// ---------------------------------------------------------------------------
__device__ __forceinline__ v16bf load_fragA(const __bf16* row, int k0, int kh) {
    v8bfv lo = *(const v8bfv*)(row + k0 + kh * 16);
    v8bfv hi = *(const v8bfv*)(row + k0 + kh * 16 + 8);
    return __builtin_shufflevector(lo, hi, 0,1,2,3,4,5,6,7,8,9,10,11,12,13,14,15);
}

__device__ __forceinline__ v8f wmma_bf16(v16bf a, v16bf b, v8f c) {
    return __builtin_amdgcn_wmma_f32_16x16x32_bf16(false, a, false, b,
                                                   (short)0, c, false, false);
}

__device__ __forceinline__ float sigmoidf_(float x) {
    return 1.0f / (1.0f + __expf(-x));
}

// ---------------------------------------------------------------------------
// Tensor Data Mover: 1D contiguous 16KB global -> LDS copy.
// D# group0: count=1 | lds_addr | global_addr(57b) | type=2
// D# group1: data_size=3 (8B units), tensor_dim0=2048, tensor_dim1=1,
//            tile_dim0=2048, tile_dim1=1, strides=2048 (one contiguous row)
// Tracked on TENSORcnt; caller issues s_wait_tensorcnt(0) before use.
// ---------------------------------------------------------------------------
__device__ __forceinline__ void tdm_copy16k(unsigned lds_addr, const void* gptr) {
    unsigned long long ga = (unsigned long long)gptr;
    v4u g0 = { 1u,                                   // count=1, is_restore=0
               lds_addr,                             // LDS byte address
               (unsigned)(ga & 0xffffffffu),         // global_addr[31:0]
               (unsigned)((ga >> 32) & 0x01ffffffu) | 0x80000000u }; // [56:32] | type=2
    v8i g1 = { (int)(3u << 16),       // data_size=3 (8 bytes)
               (int)(2048u << 16),    // tensor_dim0[15:0] in [63:48]
               (int)(1u << 16),       // tensor_dim0[31:16]=0, tensor_dim1[15:0]=1
               (int)(2048u << 16),    // tensor_dim1[31:16]=0, tile_dim0=2048
               1,                     // tile_dim1=1, tile_dim2=0
               2048,                  // tensor_dim0_stride[31:0]
               (int)(2048u << 16),    // stride0[47:32]=0, stride1[15:0]=2048
               0 };                   // stride1[47:16]=0
    v4i z = {0, 0, 0, 0};
#if __clang_major__ >= 23
    v8i z8 = {0, 0, 0, 0, 0, 0, 0, 0};
    __builtin_amdgcn_tensor_load_to_lds(g0, g1, z, z, z8, 0);
#else
    __builtin_amdgcn_tensor_load_to_lds(g0, g1, z, z, 0);
#endif
}

// ---------------------------------------------------------------------------
// K0: fp32 [K][N] -> bf16 transposed [N][K]
// ---------------------------------------------------------------------------
__global__ __launch_bounds__(256) void k_castT(const float* __restrict__ W,
                                               __bf16* __restrict__ Wt,
                                               int K, int N) {
    int i = blockIdx.x * 256 + threadIdx.x;
    if (i >= K * N) return;
    int n = i % N, k = i / N;                 // coalesced read along N
    Wt[(size_t)n * K + k] = (__bf16)W[i];
}

// ---------------------------------------------------------------------------
// K1: embedding gather -> bf16 rows [B*T][E]
// ---------------------------------------------------------------------------
__global__ __launch_bounds__(256) void k_embed(const int* __restrict__ seq,
                                               const float* __restrict__ emb,
                                               __bf16* __restrict__ X) {
    int i = blockIdx.x * 256 + threadIdx.x;   // over BT*EE
    int r = i >> 9, e = i & 511;
    X[i] = (__bf16)emb[(size_t)seq[r] * EE + e];
}

// ---------------------------------------------------------------------------
// K2: Xw = Xbf[2048x512] @ Wt^T(stored [2048][512]) + bias  -> f32 [2048][2048]
//     grid (16,16), block 256 (8 waves). Wave w: rows mbase+w*16.., 128 cols.
// ---------------------------------------------------------------------------
__global__ __launch_bounds__(256) void k_gemm_xw(const __bf16* __restrict__ A,
                                                 const __bf16* __restrict__ Bt,
                                                 const float* __restrict__ bias,
                                                 float* __restrict__ C) {
    const int tid = threadIdx.x, wave = tid >> 5, lane = tid & 31;
    const int mbase = blockIdx.y * 128 + wave * 16;
    const int nbase = blockIdx.x * 128;
    const int n15 = lane & 15, kh = lane >> 4;

    v8f acc[8];
#pragma unroll
    for (int i = 0; i < 8; ++i) acc[i] = (v8f){0,0,0,0,0,0,0,0};

    const __bf16* arow = A + (size_t)(mbase + n15) * EE;
    for (int kk = 0; kk < 16; ++kk) {
        int k0 = kk * 32;
        v16bf a = load_fragA(arow, k0, kh);
#pragma unroll
        for (int nt = 0; nt < 8; ++nt) {
            int n = nbase + nt * 16 + n15;
            v16bf b = *(const v16bf*)&Bt[(size_t)n * EE + k0 + kh * 16];
            acc[nt] = wmma_bf16(a, b, acc[nt]);
        }
    }
#pragma unroll
    for (int nt = 0; nt < 8; ++nt) {
        int n = nbase + nt * 16 + n15;
        float bb = bias[n];
#pragma unroll
        for (int r = 0; r < 8; ++r) {
            int mm = mbase + kh * 8 + r;
            C[(size_t)mm * G4H + n] = acc[nt][r] + bb;
        }
    }
}

// ---------------------------------------------------------------------------
// K3: LSTM recurrence. Persistent 16 WGs x 256 thr (8 waves).
//     WG wg owns h-indices [wg*32, wg*32+32) across all 4 gates (128 cols),
//     with its U^T slice resident in LDS (128 KB -- CDNA5 320KB/WGP), staged
//     via the Tensor Data Mover (tensor_load_to_lds + s_wait_tensorcnt).
//     Per step: z = Xw[t] + h @ U  (WMMA), gates, c/h update; h exchanged via
//     global hglob (TDM reload) + inter-WG barrier (+ cluster split-barrier).
// ---------------------------------------------------------------------------
__global__ __launch_bounds__(256) void k_lstm(const float* __restrict__ Xw,   // [BT][2048]
                                              const __bf16* __restrict__ Ut,  // [2048][512]
                                              float* __restrict__ out,        // [BT][512]
                                              __bf16* __restrict__ hglob,     // [16][512]
                                              float* __restrict__ cglob,      // [16][512]
                                              unsigned* __restrict__ bar,     // {count, gen}
                                              int T, int initFromState) {
    extern __shared__ char smem[];
    __bf16 (*Usl)[16][EE] = (__bf16(*)[16][EE])(smem);            // [8][16][512] = 128KB
    __bf16 (*hbf)[EE]     = (__bf16(*)[EE])(smem + 131072);       // [16][512]    = 16KB
    float  (*zsl)[128]    = (float(*)[128])(smem + 147456);       // [16][128]    = 8KB
    float  (*csl)[32]     = (float(*)[32])(smem + 155648);        // [16][32]     = 2KB

    const int tid = threadIdx.x, wave = tid >> 5, lane = tid & 31;
    const int wg = blockIdx.x;                // 0..15
    const int jbase = wg * 32;
    const int g = wave >> 1, s = wave & 1;    // gate, sub-tile
    const int ncol = g * 512 + jbase + s * 16;
    const int n15 = lane & 15, kh = lane >> 4;

    // Stage U^T slice via TDM: rows [ncol, ncol+16) are contiguous 16KB.
    tdm_copy16k((unsigned)(unsigned long long)&Usl[wave][0][0],
                (const void*)&Ut[(size_t)ncol * EE]);
    __builtin_amdgcn_s_wait_tensorcnt(0);

    // Init h, c
    for (int idx = tid; idx < 16 * 512; idx += 256) {
        int m = idx >> 9, j = idx & 511;
        hbf[m][j] = initFromState ? hglob[idx] : (__bf16)0.0f;
    }
    for (int idx = tid; idx < 16 * 32; idx += 256) {
        int m = idx >> 5, jl = idx & 31;
        csl[m][jl] = initFromState ? cglob[m * 512 + jbase + jl] : 0.0f;
    }
    __syncthreads();

    unsigned locgen = 0;
    if (tid == 0) locgen = *(volatile unsigned*)&bar[1];

    for (int t = 0; t < T; ++t) {
        // ---- z_partial = h @ U (16x512 @ 512x128-per-WG), one WMMA tile/wave
        v8f acc = (v8f){0,0,0,0,0,0,0,0};
        for (int kk = 0; kk < 16; ++kk) {
            int k0 = kk * 32;
            v16bf a = load_fragA(&hbf[n15][0], k0, kh);
            v16bf b = *(const v16bf*)&Usl[wave][n15][k0 + kh * 16];
            acc = wmma_bf16(a, b, acc);
        }
        // ---- z = acc + Xw[row], stash to LDS (gate-major local layout)
        {
            int colz = ncol + n15;
            int zc = g * 32 + s * 16 + n15;
#pragma unroll
            for (int r = 0; r < 8; ++r) {
                int m = kh * 8 + r;
                zsl[m][zc] = acc[r] + Xw[(size_t)(m * T + t) * G4H + colz];
            }
        }
        __syncthreads();
        // ---- gates + state update (512 (m,j) pairs / WG)
        for (int p = tid; p < 512; p += 256) {
            int m = p >> 5, jl = p & 31;
            float zi = zsl[m][jl];
            float zf = zsl[m][32 + jl];
            float zg = zsl[m][64 + jl];
            float zo = zsl[m][96 + jl];
            float ig = sigmoidf_(zi), fg = sigmoidf_(zf);
            float gg = tanhf(zg),    og = sigmoidf_(zo);
            float c = fg * csl[m][jl] + ig * gg;
            csl[m][jl] = c;
            float h = og * tanhf(c);
            int jglob = jbase + jl;
            out[(size_t)(m * T + t) * HH + jglob] = h;
            hglob[m * 512 + jglob] = (__bf16)h;
        }
        __threadfence();
        // CDNA5 cluster split-barrier (s_barrier_signal/wait -3; NOP unclustered)
        __builtin_amdgcn_s_cluster_barrier();
        // inter-WG barrier #1: h fully published
        __syncthreads();
        if (tid == 0) {
            locgen++;
            unsigned my = atomicAdd(&bar[0], 1u);
            if (my == 15u) { atomicExch(&bar[0], 0u); __threadfence(); atomicAdd(&bar[1], 1u); }
            else { while (*(volatile unsigned*)&bar[1] < locgen) __builtin_amdgcn_s_sleep(2); }
        }
        __syncthreads();
        // reload full h (bf16, 16KB contiguous) into LDS via TDM (wave 0 only)
        if (wave == 0) {
            tdm_copy16k((unsigned)(unsigned long long)&hbf[0][0], (const void*)hglob);
            __builtin_amdgcn_s_wait_tensorcnt(0);
        }
        // inter-WG barrier #2: everyone has consumed hglob before next overwrite
        __syncthreads();
        if (tid == 0) {
            locgen++;
            unsigned my = atomicAdd(&bar[0], 1u);
            if (my == 15u) { atomicExch(&bar[0], 0u); __threadfence(); atomicAdd(&bar[1], 1u); }
            else { while (*(volatile unsigned*)&bar[1] < locgen) __builtin_amdgcn_s_sleep(2); }
        }
        __syncthreads();
    }
    // persist final c (for decoder init)
    for (int p = tid; p < 512; p += 256) {
        int m = p >> 5, jl = p & 31;
        cglob[m * 512 + jbase + jl] = csl[m][jl];
    }
}

// ---------------------------------------------------------------------------
// K4: additive attention + context. grid (16 batches, 8 q-blocks of 16).
//     enc[b] block (128x512, stride-padded to 516) resident in LDS.
// ---------------------------------------------------------------------------
#define ESTR 516
__global__ __launch_bounds__(256) void k_attn(const float* __restrict__ enc,
                                              const float* __restrict__ dec,
                                              const float* __restrict__ scale,
                                              float* __restrict__ ctx) {
    extern __shared__ char smem[];
    float* encb  = (float*)smem;                     // [128][ESTR] = 264192 B
    float* sc    = (float*)(smem + 264192);          // [512]
    float* drow  = (float*)(smem + 266240);          // [512]
    float* attnw = (float*)(smem + 268288);          // [128]
    float* tmp   = (float*)(smem + 268800);          // [128]
    float* red   = (float*)(smem + 269312);          // [256]

    const int b = blockIdx.x, qb = blockIdx.y;
    const int tid = threadIdx.x;

    for (int idx = tid; idx < 128 * 128; idx += 256) {
        int k = idx >> 7, h4 = (idx & 127) * 4;
        *(float4*)&encb[k * ESTR + h4] = *(const float4*)&enc[(size_t)(b * TT + k) * HH + h4];
    }
    for (int h = tid; h < 512; h += 256) sc[h] = scale[h];
    __syncthreads();

    for (int qi = 0; qi < 16; ++qi) {
        int q = qb * 16 + qi;
        for (int h = tid; h < 512; h += 256)
            drow[h] = dec[(size_t)(b * TT + q) * HH + h];
        __syncthreads();
        // scores: 128 k x 2 h-halves
        int k = tid & 127, half = tid >> 7;
        float p = 0.f;
        for (int h = half * 256; h < half * 256 + 256; ++h)
            p += sc[h] * tanhf(drow[h] + encb[k * ESTR + h]);
        red[tid] = p;
        __syncthreads();
        if (tid < 128) { float v = red[tid] + red[tid + 128]; attnw[tid] = v; tmp[tid] = v; }
        __syncthreads();
        for (int off = 64; off > 0; off >>= 1) {
            if (tid < off) tmp[tid] = fmaxf(tmp[tid], tmp[tid + off]);
            __syncthreads();
        }
        float mx = tmp[0];
        __syncthreads();
        if (tid < 128) { float e = __expf(attnw[tid] - mx); attnw[tid] = e; tmp[tid] = e; }
        __syncthreads();
        for (int off = 64; off > 0; off >>= 1) {
            if (tid < off) tmp[tid] += tmp[tid + off];
            __syncthreads();
        }
        float inv = 1.0f / tmp[0];
        __syncthreads();
        for (int h = tid; h < 512; h += 256) {
            float a0 = 0.f;
            for (int k2 = 0; k2 < 128; ++k2) a0 += attnw[k2] * encb[k2 * ESTR + h];
            ctx[(size_t)(b * TT + q) * HH + h] = a0 * inv;
        }
        __syncthreads();
    }
}

// ---------------------------------------------------------------------------
// K5: logits = ctx[2048x512] @ W_out[512x32000] + b_out  (WMMA, W tile via LDS
//     with inline fp32->bf16 transpose; W_out lives in L2 (65MB < 192MB)).
//     grid 128 (16 rows each), block 256 (8 waves x 2 n-tiles = 256 cols/iter).
// ---------------------------------------------------------------------------
__global__ __launch_bounds__(256) void k_out(const float* __restrict__ ctx,
                                             const float* __restrict__ Wout,
                                             const float* __restrict__ bout,
                                             float* __restrict__ logits) {
    extern __shared__ char smem[];
    __bf16 (*Abf)[EE] = (__bf16(*)[EE])smem;                 // [16][512] = 16KB
    __bf16 (*Wt)[32]  = (__bf16(*)[32])(smem + 16384);       // [256][32] = 16KB

    const int tid = threadIdx.x, wave = tid >> 5, lane = tid & 31;
    const int mrow = blockIdx.x * 16;
    const int n15 = lane & 15, kh = lane >> 4;

    for (int idx = tid; idx < 16 * 512; idx += 256) {
        int m = idx >> 9, e = idx & 511;
        Abf[m][e] = (__bf16)ctx[(size_t)(mrow + m) * HH + e];
    }
    __syncthreads();

    for (int nb = 0; nb < 125; ++nb) {
        int nbase = nb * 256;
        v8f acc0 = (v8f){0,0,0,0,0,0,0,0};
        v8f acc1 = (v8f){0,0,0,0,0,0,0,0};
        for (int kk = 0; kk < 16; ++kk) {
            int k0 = kk * 32;
            __syncthreads();
            {   // stage/transposed-convert W tile: thread t owns column nbase+t
                const float* src = &Wout[(size_t)k0 * VV + nbase + tid];
                __builtin_prefetch((const void*)(src + (size_t)32 * VV), 0, 1);
#pragma unroll
                for (int kl = 0; kl < 32; ++kl)
                    Wt[tid][kl] = (__bf16)src[(size_t)kl * VV];
            }
            __syncthreads();
            v16bf a  = load_fragA(&Abf[n15][0], k0, kh);
            v16bf b0 = *(const v16bf*)&Wt[wave * 32 + n15][kh * 16];
            v16bf b1 = *(const v16bf*)&Wt[wave * 32 + 16 + n15][kh * 16];
            acc0 = wmma_bf16(a, b0, acc0);
            acc1 = wmma_bf16(a, b1, acc1);
        }
#pragma unroll
        for (int half = 0; half < 2; ++half) {
            v8f acc = half ? acc1 : acc0;
            int col = nbase + wave * 32 + half * 16 + n15;
            float bb = bout[col];
#pragma unroll
            for (int r = 0; r < 8; ++r) {
                int m = mrow + kh * 8 + r;
                logits[(size_t)m * VV + col] = acc[r] + bb;
            }
        }
    }
}

// ---------------------------------------------------------------------------
// K6: in-place row softmax over 32000 (one WG / row; streaming, L2-friendly)
// ---------------------------------------------------------------------------
__global__ __launch_bounds__(256) void k_softmax(float* __restrict__ logits) {
    __shared__ float red[256];
    const int row = blockIdx.x, tid = threadIdx.x;
    float* p = logits + (size_t)row * VV;
    float mx = -3.4e38f;
    for (int i = tid; i < VV; i += 256) mx = fmaxf(mx, p[i]);
    red[tid] = mx; __syncthreads();
    for (int off = 128; off > 0; off >>= 1) {
        if (tid < off) red[tid] = fmaxf(red[tid], red[tid + off]);
        __syncthreads();
    }
    mx = red[0]; __syncthreads();
    float s = 0.f;
    for (int i = tid; i < VV; i += 256) s += __expf(p[i] - mx);
    red[tid] = s; __syncthreads();
    for (int off = 128; off > 0; off >>= 1) {
        if (tid < off) red[tid] += red[tid + off];
        __syncthreads();
    }
    float inv = 1.0f / red[0];
    for (int i = tid; i < VV; i += 256) p[i] = __expf(p[i] - mx) * inv;
}

// ---------------------------------------------------------------------------
// Host launcher
// ---------------------------------------------------------------------------
extern "C" void kernel_launch(void* const* d_in, const int* in_sizes, int n_in,
                              void* d_out, int out_size, void* d_ws, size_t ws_size,
                              hipStream_t stream) {
    (void)in_sizes; (void)n_in; (void)out_size; (void)ws_size;
    const int*   input_seq  = (const int*)d_in[0];
    const int*   output_seq = (const int*)d_in[1];
    const float* enc_emb    = (const float*)d_in[2];
    const float* dec_emb    = (const float*)d_in[3];
    const float* W_enc      = (const float*)d_in[4];
    const float* U_enc      = (const float*)d_in[5];
    const float* b_enc      = (const float*)d_in[6];
    const float* W_dec      = (const float*)d_in[7];
    const float* U_dec      = (const float*)d_in[8];
    const float* b_dec      = (const float*)d_in[9];
    const float* attn_scale = (const float*)d_in[10];
    const float* W_out      = (const float*)d_in[11];
    const float* b_out      = (const float*)d_in[12];

    char* ws = (char*)d_ws;
    constexpr size_t MB = 1u << 20;
    __bf16* wt_enc  = (__bf16*)(ws + 0 * MB);
    __bf16* ut_enc  = (__bf16*)(ws + 2 * MB);
    __bf16* wt_dec  = (__bf16*)(ws + 4 * MB);
    __bf16* ut_dec  = (__bf16*)(ws + 6 * MB);
    __bf16* xbf_enc = (__bf16*)(ws + 8 * MB);
    __bf16* xbf_dec = (__bf16*)(ws + 10 * MB);
    float*  xw_enc  = (float*)(ws + 12 * MB);
    float*  xw_dec  = (float*)(ws + 28 * MB);
    float*  enc_out = (float*)(ws + 44 * MB);
    float*  dec_out = (float*)(ws + 48 * MB);
    float*  ctx     = (float*)(ws + 52 * MB);
    __bf16* hglob   = (__bf16*)(ws + 56 * MB);
    float*  cglob   = (float*)(ws + 56 * MB + 65536);
    unsigned* bar   = (unsigned*)(ws + 56 * MB + 131072);

    hipMemsetAsync(bar, 0, 256, stream);

    // weight transpose/convert (fp32 [K][N] -> bf16 [N][K])
    const int wgT = (EE * G4H + 255) / 256;
    k_castT<<<wgT, 256, 0, stream>>>(W_enc, wt_enc, EE, G4H);
    k_castT<<<wgT, 256, 0, stream>>>(U_enc, ut_enc, HH, G4H);
    k_castT<<<wgT, 256, 0, stream>>>(W_dec, wt_dec, EE, G4H);
    k_castT<<<wgT, 256, 0, stream>>>(U_dec, ut_dec, HH, G4H);

    // embedding gathers
    k_embed<<<(BT * EE) / 256, 256, 0, stream>>>(input_seq,  enc_emb, xbf_enc);
    k_embed<<<(BT * EE) / 256, 256, 0, stream>>>(output_seq, dec_emb, xbf_dec);

    // hoisted X@W + b pre-GEMMs
    k_gemm_xw<<<dim3(16, 16), 256, 0, stream>>>(xbf_enc, wt_enc, b_enc, xw_enc);
    k_gemm_xw<<<dim3(16, 16), 256, 0, stream>>>(xbf_dec, wt_dec, b_dec, xw_dec);

    // recurrences (persistent 16-WG kernels, U^T LDS-resident via TDM)
    const size_t SH_LSTM = 157696;
    k_lstm<<<16, 256, SH_LSTM, stream>>>(xw_enc, ut_enc, enc_out, hglob, cglob, bar, TT, 0);
    k_lstm<<<16, 256, SH_LSTM, stream>>>(xw_dec, ut_dec, dec_out, hglob, cglob, bar, TT, 1);

    // additive attention + context
    const size_t SH_ATTN = 270336;
    k_attn<<<dim3(16, 8), 256, SH_ATTN, stream>>>(enc_out, dec_out, attn_scale, ctx);

    // output projection (logits into d_out), then in-place softmax
    const size_t SH_OUT = 32768;
    k_out<<<128, 256, SH_OUT, stream>>>(ctx, W_out, b_out, (float*)d_out);
    k_softmax<<<BT, 256, 0, stream>>>((float*)d_out);
}
